// BaseModel_53455162966557
// MI455X (gfx1250) — matrix operand order repaired and checked by reading, not compiled, not run on hardware
//
#include <hip/hip_runtime.h>

// ---------------- problem constants ----------------
#define BB   2048
#define OBSN 64
#define CTXN 16
#define DDIM 256
#define MDIM 256
#define HDIM 256
#define NPAR 131584            // D*M + M + M*D + D
#define NCHUNK 514             // NPAR / 256

typedef __attribute__((ext_vector_type(16))) __bf16          v16bf;
typedef __attribute__((ext_vector_type(8)))  float           v8f;

union BFVec { uint4 q[2]; v16bf v; };
union BFU   { __bf16 h; unsigned short u; };

__device__ inline unsigned short f2bf(float f) {
  BFU b; b.h = (__bf16)f;      // native v_cvt to bf16
  return b.u;
}

__device__ inline v8f v8zero() {
  v8f z = {0.f,0.f,0.f,0.f,0.f,0.f,0.f,0.f};
  return z;
}

__device__ inline v8f wmma_bf16(v16bf a, v16bf b, v8f c) {
  // D = A(16x32 bf16) * B(32x16 bf16) + C(16x16 f32)
  return __builtin_amdgcn_wmma_f32_16x16x32_bf16(false, a, false, b, (short)0, c,
                                                 false, false);
}

// A-fragment from LDS bf16 row-major [rows][ldk].
// lane<16 -> row=lane, K {0..15}; lane>=16 -> row=lane-16, K {16..31}
__device__ inline v16bf load_afrag_lds(const unsigned short* S, int ldk, int row0, int kbase) {
  int l  = threadIdx.x & 31;
  const unsigned short* p = S + (size_t)(row0 + (l & 15)) * ldk + kbase + ((l >> 4) << 4);
  BFVec u;
  u.q[0] = *(const uint4*)p;          // K  .. K+7
  u.q[1] = *(const uint4*)(p + 8);    // K+8 .. K+15
  return u.v;
}

// B-fragment from fragment-packed bf16 buffer: 32 lanes x 32 bytes contiguous (1KB/fragment)
__device__ inline v16bf load_bfrag_packed(const unsigned short* base) {
  int l = threadIdx.x & 31;
  const unsigned short* p = base + l * 16;
  BFVec u;
  u.q[0] = *(const uint4*)p;
  u.q[1] = *(const uint4*)(p + 8);
  return u.v;
}

// ------------------------------------------------------------------
// Kernel 1: repack Wh2 (H x NPAR fp32) -> bf16 WMMA-B fragment-major.
// Fragment f = (chunk*16 + ntile)*8 + kblock holds a 32(K) x 16(N) tile,
// laid out as lane*32B so the main loop issues coalesced b128 loads.
// grid = NCHUNK*16 blocks of 256.
// ------------------------------------------------------------------
__global__ void __launch_bounds__(256) wh2_pack_kernel(const float* __restrict__ Wh2,
                                                       unsigned short* __restrict__ wh2f) {
  __shared__ unsigned short sT[16 * 256];          // [p(col)][k] transposed staging
  int blk = blockIdx.x;
  int tid = threadIdx.x;
  size_t p0 = (size_t)blk * 16;                    // 16 original columns per block
  {
    int k = tid;                                   // 256 threads == H rows
    const float* src = Wh2 + (size_t)k * NPAR + p0;
    float4 f0 = ((const float4*)src)[0];
    float4 f1 = ((const float4*)src)[1];
    float4 f2 = ((const float4*)src)[2];
    float4 f3 = ((const float4*)src)[3];
    float vals[16] = {f0.x,f0.y,f0.z,f0.w, f1.x,f1.y,f1.z,f1.w,
                      f2.x,f2.y,f2.z,f2.w, f3.x,f3.y,f3.z,f3.w};
    #pragma unroll
    for (int p = 0; p < 16; ++p) sT[p * 256 + k] = f2bf(vals[p]);
  }
  __syncthreads();
  {
    int kb = tid >> 5, l = tid & 31;               // 8 kblocks x 32 lanes
    const unsigned short* src = &sT[(l & 15) * 256 + kb * 32 + ((l >> 4) << 4)];
    unsigned short* dst = wh2f + ((size_t)blk * 8 + kb) * 512 + l * 16;
    ((uint4*)dst)[0] = ((const uint4*)src)[0];
    ((uint4*)dst)[1] = ((const uint4*)src)[1];
  }
}

// ------------------------------------------------------------------
// Kernel 1b: pack a small fp32 weight matrix (kmax x 256, row stride ld)
// into B-fragment-major bf16, zero-padded in K up to kbCount*32.
// grid = 16 blocks (one per ntile) x 256 threads.
// Done ONCE so the 128 prep blocks never re-gather/re-convert weights.
// ------------------------------------------------------------------
__global__ void __launch_bounds__(256) packB_kernel(const float* __restrict__ M, int kmax,
                                                    int kbCount, int ld,
                                                    unsigned short* __restrict__ dst) {
  __shared__ unsigned short sT[16 * 256];
  int nt  = blockIdx.x;
  int tid = threadIdx.x;
  int k   = tid;
  #pragma unroll
  for (int p = 0; p < 16; ++p) {
    float f = (k < kmax) ? M[(size_t)k * ld + nt * 16 + p] : 0.f;
    sT[p * 256 + k] = f2bf(f);
  }
  __syncthreads();
  int kb = tid >> 5, l = tid & 31;
  if (kb < kbCount) {
    const unsigned short* src = &sT[(l & 15) * 256 + kb * 32 + ((l >> 4) << 4)];
    unsigned short* d = dst + ((size_t)nt * kbCount + kb) * 512 + l * 16;
    ((uint4*)d)[0] = ((const uint4*)src)[0];
    ((uint4*)d)[1] = ((const uint4*)src)[1];
  }
}

// ------------------------------------------------------------------
// Kernel 2: h = relu(relu(x@W0+b0)@W1+b1), c = relu(ctx@Wh1+bh1), via WMMA.
// All weights come pre-packed bf16 (w0f/w1f/wh1f) -> pure b128 loads.
// Outputs: hT  [B/32][256 d][32 b] fp32  (transposed for broadcast loads)
//          cB  [B/32][32 b][256 k] bf16  (A-fragment friendly row-major)
// grid = B/16 blocks of 256 (8 waves x 2 ntiles cover N=256).
// ------------------------------------------------------------------
__global__ void __launch_bounds__(256) prep_hc_kernel(
    const float* __restrict__ x, const float* __restrict__ ctx,
    const unsigned short* __restrict__ w0f, const float* __restrict__ b0,
    const unsigned short* __restrict__ w1f, const float* __restrict__ b1,
    const unsigned short* __restrict__ wh1f, const float* __restrict__ bh1,
    float* __restrict__ hT, unsigned short* __restrict__ cB) {
  __shared__ unsigned short sX[16 * 64];
  __shared__ unsigned short sCtx[16 * 32];
  __shared__ unsigned short sH1[16 * 256];
  int pb  = blockIdx.x;
  int tid = threadIdx.x;
  int bb  = pb * 16;

  for (int i = tid; i < 16 * 64; i += 256) {
    int r = i >> 6, k = i & 63;
    sX[i] = f2bf(x[(size_t)(bb + r) * OBSN + k]);
  }
  for (int i = tid; i < 16 * 32; i += 256) {
    int r = i >> 5, k = i & 31;
    sCtx[i] = f2bf((k < CTXN) ? ctx[(size_t)(bb + r) * CTXN + k] : 0.f);
  }
  __syncthreads();

  int wave = tid >> 5, lane = tid & 31, ln = lane & 15, lh = lane >> 4;

  // GEMM1: h1 = relu(x @ W0 + b0)   (16 x 64) @ (64 x 256)
  #pragma unroll
  for (int nt = 0; nt < 2; ++nt) {
    int ntg = wave * 2 + nt;
    int n   = ntg * 16 + ln;
    v8f acc = v8zero();
    #pragma unroll
    for (int kb = 0; kb < 2; ++kb) {
      v16bf a  = load_afrag_lds(sX, 64, 0, kb * 32);
      v16bf bm = load_bfrag_packed(w0f + ((size_t)ntg * 2 + kb) * 512);
      acc = wmma_bf16(a, bm, acc);
    }
    float bias = b0[n];
    #pragma unroll
    for (int v = 0; v < 8; ++v) {
      float hv = fmaxf(acc[v] + bias, 0.f);
      sH1[(size_t)(lh * 8 + v) * 256 + n] = f2bf(hv);
    }
  }
  __syncthreads();

  // GEMM2: h = relu(h1 @ W1 + b1)   (16 x 256) @ (256 x 256) -> hT (fp32 transposed)
  #pragma unroll
  for (int nt = 0; nt < 2; ++nt) {
    int ntg = wave * 2 + nt;
    int n   = ntg * 16 + ln;
    v8f acc = v8zero();
    #pragma unroll
    for (int kb = 0; kb < 8; ++kb) {
      v16bf a  = load_afrag_lds(sH1, 256, 0, kb * 32);
      v16bf bm = load_bfrag_packed(w1f + ((size_t)ntg * 8 + kb) * 512);
      acc = wmma_bf16(a, bm, acc);
    }
    float bias = b1[n];
    float4 o0, o1;
    o0.x = fmaxf(acc[0] + bias, 0.f); o0.y = fmaxf(acc[1] + bias, 0.f);
    o0.z = fmaxf(acc[2] + bias, 0.f); o0.w = fmaxf(acc[3] + bias, 0.f);
    o1.x = fmaxf(acc[4] + bias, 0.f); o1.y = fmaxf(acc[5] + bias, 0.f);
    o1.z = fmaxf(acc[6] + bias, 0.f); o1.w = fmaxf(acc[7] + bias, 0.f);
    float* dst = hT + ((size_t)(pb >> 1) * 256 + n) * 32 + (pb & 1) * 16 + lh * 8;
    ((float4*)dst)[0] = o0;
    ((float4*)dst)[1] = o1;
  }

  // GEMM3: c = relu(ctx @ Wh1 + bh1)   (16 x 16pad32) @ (16 x 256) -> cB (bf16 row-major)
  #pragma unroll
  for (int nt = 0; nt < 2; ++nt) {
    int ntg = wave * 2 + nt;
    int n   = ntg * 16 + ln;
    v16bf a  = load_afrag_lds(sCtx, 32, 0, 0);
    v16bf bm = load_bfrag_packed(wh1f + (size_t)ntg * 512);
    v8f acc  = wmma_bf16(a, bm, v8zero());
    float bias = bh1[n];
    #pragma unroll
    for (int v = 0; v < 8; ++v) {
      float cv = fmaxf(acc[v] + bias, 0.f);
      cB[((size_t)(pb >> 1) * 32 + (pb & 1) * 16 + lh * 8 + v) * 256 + n] = f2bf(cv);
    }
  }
}

// ------------------------------------------------------------------
// Kernel 3: fused hypernetwork. Per 32-batch tile, stream all 514
// 256-column chunks of flat = c@Wh2+bh2 through WMMA without ever
// materializing flat; contract on the fly:
//   phase1 (chunks 0..255):  z1[b,m] += h[b,d] * chunk_d[b,m]; chunk 256 = b1m
//   phase2 (chunks 257..512): z2[b,d] += z[b,m] * chunk_m[b,d]; chunk 513 = b2m
//   out = relu(h + z2)
// 64 blocks x 256 threads (8 waves; wave -> 2 ntiles x 2 mtiles).
// A-fragments (c tile) live in registers for the whole kernel.
// Next chunk's 16KB B-region is software-prefetched (global_prefetch_b8).
// ------------------------------------------------------------------
__global__ void __launch_bounds__(256) hyper_main_kernel(
    const unsigned short* __restrict__ wh2f, const float* __restrict__ bh2,
    const float* __restrict__ hT, const unsigned short* __restrict__ cB,
    float* __restrict__ out) {
  extern __shared__ char smem[];
  unsigned short* sC  = (unsigned short*)smem;            // [32][256] bf16  16KB
  float*          sHT = (float*)(smem + 32 * 256 * 2);    // [256][32] f32   32KB
  float*          sZT = sHT + 256 * 32;                   // [256][32] f32   32KB

  int tile = blockIdx.x, tid = threadIdx.x;

  const unsigned short* cs = cB + (size_t)tile * 32 * 256;
  for (int i = tid; i < 32 * 256 / 8; i += 256) ((uint4*)sC)[i] = ((const uint4*)cs)[i];
  const float* hs = hT + (size_t)tile * 256 * 32;
  for (int i = tid; i < 256 * 32 / 4; i += 256) ((float4*)sHT)[i] = ((const float4*)hs)[i];
  __syncthreads();

  int wave = tid >> 5, lane = tid & 31, ln = lane & 15, lh = lane >> 4;

  // c-tile A fragments: constant across all 514 chunks -> keep in VGPRs
  v16bf A[2][8];
  #pragma unroll
  for (int mt = 0; mt < 2; ++mt)
    #pragma unroll
    for (int kb = 0; kb < 8; ++kb)
      A[mt][kb] = load_afrag_lds(sC, 256, mt * 16, kb * 32);

  // ---------------- phase 1 ----------------
  v8f z1[2][2];
  #pragma unroll
  for (int mt = 0; mt < 2; ++mt)
    #pragma unroll
    for (int nt = 0; nt < 2; ++nt) z1[mt][nt] = v8zero();

  for (int cid = 0; cid <= 256; ++cid) {
    // prefetch next chunk's B region for this wave (16KB, 128B per lane x4)
    {
      const char* nf = (const char*)(wh2f + (((size_t)(cid + 1) * 16 + wave * 2) * 8) * 512);
      #pragma unroll
      for (int pf = 0; pf < 4; ++pf)
        __builtin_prefetch(nf + pf * 4096 + lane * 128, 0, 1);
    }
    v8f Dm[2][2];
    #pragma unroll
    for (int mt = 0; mt < 2; ++mt)
      #pragma unroll
      for (int nt = 0; nt < 2; ++nt) Dm[mt][nt] = v8zero();
    #pragma unroll
    for (int kb = 0; kb < 8; ++kb) {
      v16bf Bf[2];
      #pragma unroll
      for (int nt = 0; nt < 2; ++nt)
        Bf[nt] = load_bfrag_packed(wh2f + (((size_t)cid * 16 + wave * 2 + nt) * 8 + kb) * 512);
      #pragma unroll
      for (int mt = 0; mt < 2; ++mt)
        #pragma unroll
        for (int nt = 0; nt < 2; ++nt)
          Dm[mt][nt] = wmma_bf16(A[mt][kb], Bf[nt], Dm[mt][nt]);
    }
    #pragma unroll
    for (int nt = 0; nt < 2; ++nt) {
      float bias = bh2[(size_t)cid * 256 + wave * 32 + nt * 16 + ln];
      #pragma unroll
      for (int mt = 0; mt < 2; ++mt)
        #pragma unroll
        for (int v = 0; v < 8; ++v) Dm[mt][nt][v] += bias;
    }
    if (cid < 256) {
      #pragma unroll
      for (int mt = 0; mt < 2; ++mt) {
        const float* hp = &sHT[cid * 32 + mt * 16 + lh * 8];
        float4 h0 = ((const float4*)hp)[0], h1 = ((const float4*)hp)[1];
        float hv[8] = {h0.x,h0.y,h0.z,h0.w, h1.x,h1.y,h1.z,h1.w};
        #pragma unroll
        for (int nt = 0; nt < 2; ++nt)
          #pragma unroll
          for (int v = 0; v < 8; ++v) z1[mt][nt][v] += hv[v] * Dm[mt][nt][v];
      }
    } else {   // b1m bias chunk
      #pragma unroll
      for (int mt = 0; mt < 2; ++mt)
        #pragma unroll
        for (int nt = 0; nt < 2; ++nt) z1[mt][nt] += Dm[mt][nt];
    }
  }

  // relu(z1) -> LDS transposed [m][b]
  #pragma unroll
  for (int mt = 0; mt < 2; ++mt)
    #pragma unroll
    for (int nt = 0; nt < 2; ++nt) {
      int m = wave * 32 + nt * 16 + ln;
      #pragma unroll
      for (int v = 0; v < 8; ++v)
        sZT[(size_t)m * 32 + mt * 16 + lh * 8 + v] = fmaxf(z1[mt][nt][v], 0.f);
    }
  __syncthreads();

  // ---------------- phase 2 ----------------
  v8f z2[2][2];
  #pragma unroll
  for (int mt = 0; mt < 2; ++mt)
    #pragma unroll
    for (int nt = 0; nt < 2; ++nt) z2[mt][nt] = v8zero();

  for (int mc = 0; mc <= 256; ++mc) {
    int cid = 257 + mc;
    if (mc < 256) {
      const char* nf = (const char*)(wh2f + (((size_t)(cid + 1) * 16 + wave * 2) * 8) * 512);
      #pragma unroll
      for (int pf = 0; pf < 4; ++pf)
        __builtin_prefetch(nf + pf * 4096 + lane * 128, 0, 1);
    }
    v8f Dm[2][2];
    #pragma unroll
    for (int mt = 0; mt < 2; ++mt)
      #pragma unroll
      for (int nt = 0; nt < 2; ++nt) Dm[mt][nt] = v8zero();
    #pragma unroll
    for (int kb = 0; kb < 8; ++kb) {
      v16bf Bf[2];
      #pragma unroll
      for (int nt = 0; nt < 2; ++nt)
        Bf[nt] = load_bfrag_packed(wh2f + (((size_t)cid * 16 + wave * 2 + nt) * 8 + kb) * 512);
      #pragma unroll
      for (int mt = 0; mt < 2; ++mt)
        #pragma unroll
        for (int nt = 0; nt < 2; ++nt)
          Dm[mt][nt] = wmma_bf16(A[mt][kb], Bf[nt], Dm[mt][nt]);
    }
    #pragma unroll
    for (int nt = 0; nt < 2; ++nt) {
      float bias = bh2[(size_t)cid * 256 + wave * 32 + nt * 16 + ln];
      #pragma unroll
      for (int mt = 0; mt < 2; ++mt)
        #pragma unroll
        for (int v = 0; v < 8; ++v) Dm[mt][nt][v] += bias;
    }
    if (mc < 256) {
      #pragma unroll
      for (int mt = 0; mt < 2; ++mt) {
        const float* zp = &sZT[mc * 32 + mt * 16 + lh * 8];
        float4 q0 = ((const float4*)zp)[0], q1 = ((const float4*)zp)[1];
        float zv[8] = {q0.x,q0.y,q0.z,q0.w, q1.x,q1.y,q1.z,q1.w};
        #pragma unroll
        for (int nt = 0; nt < 2; ++nt)
          #pragma unroll
          for (int v = 0; v < 8; ++v) z2[mt][nt][v] += zv[v] * Dm[mt][nt][v];
      }
    } else {   // b2m bias chunk
      #pragma unroll
      for (int mt = 0; mt < 2; ++mt)
        #pragma unroll
        for (int nt = 0; nt < 2; ++nt) z2[mt][nt] += Dm[mt][nt];
    }
  }

  // out = relu(h + z2)
  #pragma unroll
  for (int mt = 0; mt < 2; ++mt)
    #pragma unroll
    for (int nt = 0; nt < 2; ++nt) {
      int d = wave * 32 + nt * 16 + ln;
      const float* hp = &sHT[(size_t)d * 32 + mt * 16 + lh * 8];
      float4 h0 = ((const float4*)hp)[0], h1 = ((const float4*)hp)[1];
      float hv[8] = {h0.x,h0.y,h0.z,h0.w, h1.x,h1.y,h1.z,h1.w};
      int brow = tile * 32 + mt * 16 + lh * 8;
      #pragma unroll
      for (int v = 0; v < 8; ++v)
        out[(size_t)(brow + v) * 256 + d] = fmaxf(hv[v] + z2[mt][nt][v], 0.f);
    }
}

// ------------------------------------------------------------------
extern "C" void kernel_launch(void* const* d_in, const int* in_sizes, int n_in,
                              void* d_out, int out_size, void* d_ws, size_t ws_size,
                              hipStream_t stream) {
  const float* x   = (const float*)d_in[0];
  const float* ctx = (const float*)d_in[1];
  const float* W0  = (const float*)d_in[2];
  const float* b0  = (const float*)d_in[3];
  const float* W1  = (const float*)d_in[4];
  const float* b1  = (const float*)d_in[5];
  const float* Wh1 = (const float*)d_in[6];
  const float* bh1 = (const float*)d_in[7];
  const float* Wh2 = (const float*)d_in[8];
  const float* bh2 = (const float*)d_in[9];
  float* out = (float*)d_out;

  // workspace layout (~70.7 MB):
  //   wh2f: NCHUNK*16*8 fragments * 1KB = 67,371,008 B (bf16 packed Wh2, L2-resident)
  //   hT  : 2,097,152 B   cB : 1,048,576 B
  //   w0f : 32,768 B      w1f: 131,072 B     wh1f: 16,384 B
  char* ws = (char*)d_ws;
  unsigned short* wh2f = (unsigned short*)ws;
  float*          hT   = (float*)(ws + 67371008);
  unsigned short* cB   = (unsigned short*)(ws + 67371008 + 2097152);
  unsigned short* w0f  = (unsigned short*)(ws + 70516736);
  unsigned short* w1f  = (unsigned short*)(ws + 70516736 + 32768);
  unsigned short* wh1f = (unsigned short*)(ws + 70516736 + 32768 + 131072);

  wh2_pack_kernel<<<NCHUNK * 16, 256, 0, stream>>>(Wh2, wh2f);
  packB_kernel<<<16, 256, 0, stream>>>(W0, OBSN, 2, DDIM, w0f);
  packB_kernel<<<16, 256, 0, stream>>>(W1, DDIM, 8, DDIM, w1f);
  packB_kernel<<<16, 256, 0, stream>>>(Wh1, CTXN, 1, HDIM, wh1f);
  prep_hc_kernel<<<BB / 16, 256, 0, stream>>>(x, ctx, w0f, b0, w1f, b1, wh1f, bh1, hT, cB);
  hyper_main_kernel<<<BB / 32, 256, 80 * 1024, stream>>>(wh2f, bh2, hT, cB, out);
}
